// TemporalDINOv3StereoModel_4277787426944
// MI455X (gfx1250) — matrix operand order repaired and checked by reading, not compiled, not run on hardware
//
#include <hip/hip_runtime.h>
#include <cstdint>
#include <cstddef>

// ============================================================================
// MI455X / gfx1250 implementation of TemporalDINOv3StereoModel forward.
// All conv / correlation math routed through v_wmma_f32_16x16x32_bf16
// (bf16 in, f32 accumulate).  Wave32, 8 waves/block, 128x128x64 tiles,
// double-buffered LDS fed by GLOBAL_LOAD_ASYNC_TO_LDS (ASYNCcnt-tracked).
// ============================================================================

typedef __attribute__((ext_vector_type(16))) __bf16 v16bf;
typedef __attribute__((ext_vector_type(8)))  __bf16 v8bf;
typedef __attribute__((ext_vector_type(8)))  float  v8f;
typedef __attribute__((ext_vector_type(4)))  int    v4i_;

union V16 { v16bf v; v8bf h[2]; };

// ---- CDNA5 async global->LDS copy (ASYNCcnt), compile-safe detection ----
#if defined(__has_builtin)
#  if __has_builtin(__builtin_amdgcn_global_load_async_to_lds_b128)
#    define HAVE_ASYNC_LDS 1
#  endif
#endif
#ifndef HAVE_ASYNC_LDS
#  define HAVE_ASYNC_LDS 0
#endif

__device__ __forceinline__ void copy16_to_lds(__bf16* lds, const __bf16* g) {
#if HAVE_ASYNC_LDS
  // prototype: (v4i addrspace(1)* src, v4i addrspace(3)* dst, imm off, imm cpol)
  // generic -> AS1/AS3 conversions are implicit for builtin args in HIP.
  __builtin_amdgcn_global_load_async_to_lds_b128((v4i_*)g, (v4i_*)lds, 0, 0);
#else
  *(v8bf*)lds = *(const v8bf*)g;
#endif
}
__device__ __forceinline__ void wait_async_8() {
#if HAVE_ASYNC_LDS
#  if __has_builtin(__builtin_amdgcn_s_wait_asynccnt)
  __builtin_amdgcn_s_wait_asynccnt(8);
#  else
  asm volatile("s_wait_asynccnt 0x8" ::: "memory");
#  endif
#endif
}
__device__ __forceinline__ void wait_async_0() {
#if HAVE_ASYNC_LDS
#  if __has_builtin(__builtin_amdgcn_s_wait_asynccnt)
  __builtin_amdgcn_s_wait_asynccnt(0);
#  else
  asm volatile("s_wait_asynccnt 0x0" ::: "memory");
#  endif
#endif
}

__device__ __forceinline__ __bf16 f2bf(float f) {
  union { float f; uint32_t u; } v; v.f = f;
  uint32_t r = v.u + 0x7FFFu + ((v.u >> 16) & 1u);   // round-to-nearest-even
  union { uint16_t s; __bf16 b; } o; o.s = (uint16_t)(r >> 16); return o.b;
}
__device__ __forceinline__ float bf2f(__bf16 b) {
  union { uint16_t s; __bf16 b; } i; i.b = b;
  union { float f; uint32_t u; } v; v.u = ((uint32_t)i.s) << 16; return v.f;
}
__device__ __forceinline__ float act_apply(float v, int act) {
  if (act == 1) return fmaxf(v, 0.f);
  if (act == 2) return 1.f / (1.f + __expf(-v));
  if (act == 3) return tanhf(v);
  return v;
}

// ----------------------------------------------------------------------------
// Tiled bf16 WMMA GEMM.
//   A  : [M][K]  bf16 row-major (K multiple of 64, M multiple of 128)
//   Bt : [N][K]  bf16 (B transposed; rows padded to K)
//   Cf : optional f32 out, channel-major  Cf[n*M + m]   (+bias, +activation)
//   Cb : optional bf16 out, row-major     Cb[m*N + n]   (used for corr volume)
// Block tile 128x128, BK=64, double-buffered LDS (64 KB), async staging.
// 8 wave32s; wave tile 32x64 = 2x4 16x16 frags; 16 WMMAs per K-stage.
// Fragment register layouts follow CDNA5 ISA 7.12.2.
// ----------------------------------------------------------------------------
#define BM 128
#define BN 128
#define BK 64

__global__ __launch_bounds__(256, 2)
void gemm_bf16_wmma(const __bf16* __restrict__ A, const __bf16* __restrict__ Bt,
                    const float* __restrict__ bias,
                    float* __restrict__ Cf, __bf16* __restrict__ Cb,
                    int M, int N, int K, int act) {
  __shared__ __align__(16) __bf16 As[2][BM][BK];   // 2 * 16 KB
  __shared__ __align__(16) __bf16 Bs[2][BN][BK];   // 2 * 16 KB

  const int tid  = threadIdx.x;
  const int lane = tid & 31;
  const int wave = tid >> 5;
  const int wm   = wave & 3;        // m offset wm*32
  const int wn   = wave >> 2;       // n offset wn*64
  const int m0   = blockIdx.x * BM;
  const int n0   = blockIdx.y * BN;

  v8f acc[2][4];
  for (int a = 0; a < 2; ++a)
    for (int b = 0; b < 4; ++b)
      for (int q = 0; q < 8; ++q) acc[a][b][q] = 0.f;

  const int r  = lane & 15;
  const int kh = (lane >> 4) * 8;    // A: K sub-offset per ISA A-layout
  const int kb = (lane >> 4) * 16;   // unused; B is loaded with kh, same k order as A

  // stage one 128x64 A tile + 128x64 B tile into LDS buffer `buf`
  // (8 async b128 ops per thread; OOB N rows clamped to row 0 — the garbage
  //  columns are computed but never stored)
  auto stage = [&](int buf, int k0) {
    for (int i = 0; i < 4; ++i) {
      int chunk = tid + i * 256;         // 0..1023
      int row   = chunk >> 3;            // 0..127
      int cc    = (chunk & 7) * 8;       // 0,8,...,56
      copy16_to_lds(&As[buf][row][cc], &A[(size_t)(m0 + row) * K + k0 + cc]);
      int gn = n0 + row;
      int gnc = (gn < N) ? gn : 0;
      copy16_to_lds(&Bs[buf][row][cc], &Bt[(size_t)gnc * K + k0 + cc]);
#if !HAVE_ASYNC_LDS
      if (k0 + BK < K)
        __builtin_prefetch(&A[(size_t)(m0 + row) * K + k0 + BK + cc], 0, 0);
#endif
    }
  };

  const int nk = K / BK;
  stage(0, 0);
  for (int s = 0; s < nk; ++s) {
    const int buf = s & 1;
    if (s + 1 < nk) {        // kick off next tile, then wait for current one
      stage(buf ^ 1, (s + 1) * BK);
      wait_async_8();        // ASYNCcnt <= 8  ->  tile `buf` has landed
    } else {
      wait_async_0();
    }
    __syncthreads();

    for (int ks = 0; ks < BK; ks += 32) {
      V16 af[2];
      for (int mi = 0; mi < 2; ++mi) {
        int mr = wm * 32 + mi * 16 + r;
        af[mi].h[0] = *(const v8bf*)&As[buf][mr][ks + kh];
        af[mi].h[1] = *(const v8bf*)&As[buf][mr][ks + 16 + kh];
      }
      for (int ni = 0; ni < 4; ++ni) {
        V16 bf_;
        int nr = wn * 64 + ni * 16 + r;
        bf_.h[0] = *(const v8bf*)&Bs[buf][nr][ks + kh];
        bf_.h[1] = *(const v8bf*)&Bs[buf][nr][ks + 16 + kh];
        acc[0][ni] = __builtin_amdgcn_wmma_f32_16x16x32_bf16(
            false, af[0].v, false, bf_.v, (short)0, acc[0][ni], false, false);
        acc[1][ni] = __builtin_amdgcn_wmma_f32_16x16x32_bf16(
            false, af[1].v, false, bf_.v, (short)0, acc[1][ni], false, false);
      }
    }
    __syncthreads();   // all waves done reading `buf` before it is re-staged
  }

  // ---- epilogue: C/D layout: VGPR q -> M = q + (lane>=16?8:0), N = lane&15
  const int mlane = (lane >> 4) * 8;
  for (int mi = 0; mi < 2; ++mi)
    for (int ni = 0; ni < 4; ++ni) {
      int n = n0 + wn * 64 + ni * 16 + (lane & 15);
      if (n >= N) continue;
      float bv = bias ? bias[n] : 0.f;
      for (int q = 0; q < 8; ++q) {
        int m = m0 + wm * 32 + mi * 16 + q + mlane;
        float v = act_apply(acc[mi][ni][q] + bv, act);
        if (Cf) Cf[(size_t)n * M + m] = v;
        if (Cb) Cb[(size_t)m * N + n] = f2bf(v);
      }
    }
}

// ----------------------------------------------------------------------------
// Weight pack: f32 [N][K] -> bf16 [N][Kp] (zero K-pad)
// ----------------------------------------------------------------------------
__global__ void pack_w_k(const float* __restrict__ w, __bf16* __restrict__ out,
                         int N, int K, int Kp) {
  size_t idx = (size_t)blockIdx.x * blockDim.x + threadIdx.x;
  size_t tot = (size_t)N * Kp;
  if (idx >= tot) return;
  int o = (int)(idx / Kp), kk = (int)(idx % Kp);
  out[idx] = (kk < K) ? f2bf(w[(size_t)o * K + kk]) : f2bf(0.f);
}

// ----------------------------------------------------------------------------
// 1x1 pack: up-to-3 fp32 channel-major sources -> bf16 [p][Kp] (concat fused)
// ----------------------------------------------------------------------------
__global__ void pack_rows_k(const float* s0, int c0, const float* s1, int c1,
                            const float* s2, int c2, int HW, int Kp,
                            __bf16* __restrict__ out) {
  size_t idx = (size_t)blockIdx.x * blockDim.x + threadIdx.x;
  size_t tot = (size_t)HW * Kp;
  if (idx >= tot) return;
  int p = (int)(idx / Kp), c = (int)(idx % Kp);
  int C = c0 + c1 + c2;
  float v = 0.f;
  if (c < C) {
    const float* s = s0; int cl = c;
    if (cl >= c0) { cl -= c0; s = s1; if (cl >= c1) { cl -= c1; s = s2; } }
    v = s[(size_t)cl * HW + p];
  }
  out[idx] = f2bf(v);
}

// ----------------------------------------------------------------------------
// im2col 3x3 (pad=1, stride 1 or 2), up-to-3 concat sources -> bf16 [p][Kp]
// K index = c*9 + ky*3 + kx, matching OIHW weight flattening.
// ----------------------------------------------------------------------------
__global__ void im2col3x3_k(const float* s0, int c0, const float* s1, int c1,
                            const float* s2, int c2, int H, int W, int stride,
                            int Kp, __bf16* __restrict__ out) {
  int Ho = H / stride, Wo = W / stride;
  size_t idx = (size_t)blockIdx.x * blockDim.x + threadIdx.x;
  size_t tot = (size_t)Ho * Wo * Kp;
  if (idx >= tot) return;
  int p = (int)(idx / Kp), kk = (int)(idx % Kp);
  int C = c0 + c1 + c2;
  float v = 0.f;
  if (kk < C * 9) {
    int c = kk / 9, j = kk % 9, ky = j / 3, kx = j % 3;
    int oy = p / Wo, ox = p % Wo;
    int iy = oy * stride + ky - 1, ix = ox * stride + kx - 1;
    if (iy >= 0 && iy < H && ix >= 0 && ix < W) {
      const float* s = s0; int cl = c;
      if (cl >= c0) { cl -= c0; s = s1; if (cl >= c1) { cl -= c1; s = s2; } }
      v = s[(size_t)cl * H * W + (size_t)iy * W + ix];
    }
  }
  out[idx] = f2bf(v);
}

// ----------------------------------------------------------------------------
// GroupNorm (groups=8) over channel-major f32, optional fused relu. In-place ok.
// ----------------------------------------------------------------------------
__global__ void groupnorm_k(const float* __restrict__ x, float* __restrict__ y,
                            const float* __restrict__ gamma,
                            const float* __restrict__ beta,
                            int C, int HW, int act) {
  const int grp = blockIdx.x;
  const int cg = C / 8;
  const size_t base = (size_t)grp * cg * HW;
  const size_t n = (size_t)cg * HW;
  float s = 0.f, ss = 0.f;
  for (size_t i = threadIdx.x; i < n; i += blockDim.x) {
    float v = x[base + i]; s += v; ss += v * v;
  }
  __shared__ float rs[256], rss[256];
  rs[threadIdx.x] = s; rss[threadIdx.x] = ss;
  __syncthreads();
  for (int o = 128; o > 0; o >>= 1) {
    if ((int)threadIdx.x < o) {
      rs[threadIdx.x] += rs[threadIdx.x + o];
      rss[threadIdx.x] += rss[threadIdx.x + o];
    }
    __syncthreads();
  }
  float mean = rs[0] / (float)n;
  float var = rss[0] / (float)n - mean * mean;
  float inv = rsqrtf(var + 1e-5f);
  for (size_t i = threadIdx.x; i < n; i += blockDim.x) {
    int c = grp * cg + (int)(i / HW);
    float v = (x[base + i] - mean) * inv * gamma[c] + beta[c];
    if (act == 1) v = fmaxf(v, 0.f);
    y[base + i] = v;
  }
}

// ----------------------------------------------------------------------------
// Per-pixel l2norm over C channels + transpose to pixel-major bf16 [p][C].
// ----------------------------------------------------------------------------
__global__ void l2norm_pack_k(const float* __restrict__ x,
                              __bf16* __restrict__ out, int C, int HW) {
  int p = blockIdx.x * blockDim.x + threadIdx.x;
  if (p >= HW) return;
  float s = 0.f;
  for (int c = 0; c < C; ++c) { float v = x[(size_t)c * HW + p]; s += v * v; }
  float inv = 1.f / fmaxf(sqrtf(s), 1e-12f);
  for (int c = 0; c < C; ++c)
    out[(size_t)p * C + c] = f2bf(x[(size_t)c * HW + p] * inv);
}

// ----------------------------------------------------------------------------
// 2x2 avg-pool, 8192 independent bf16 images (pyramid construction)
// ----------------------------------------------------------------------------
__global__ void avgpool2_k(const __bf16* __restrict__ in,
                           __bf16* __restrict__ out, int Hi, int Wi) {
  int Ho = Hi / 2, Wo = Wi / 2;
  size_t tot = (size_t)8192 * Ho * Wo;
  size_t idx = (size_t)blockIdx.x * blockDim.x + threadIdx.x;
  if (idx >= tot) return;
  size_t img = idx / ((size_t)Ho * Wo);
  int q = (int)(idx % ((size_t)Ho * Wo));
  int oy = q / Wo, ox = q % Wo;
  const __bf16* im = in + img * (size_t)Hi * Wi;
  float v = 0.25f * (bf2f(im[(2 * oy) * Wi + 2 * ox]) +
                     bf2f(im[(2 * oy) * Wi + 2 * ox + 1]) +
                     bf2f(im[(2 * oy + 1) * Wi + 2 * ox]) +
                     bf2f(im[(2 * oy + 1) * Wi + 2 * ox + 1]));
  out[idx] = f2bf(v);
}

// ----------------------------------------------------------------------------
// Correlation lookup (mirrors the reference math exactly, including its
// delta/(x,y) component pairing and normalized grid_sample with border clamp).
// Writes channels 2..325 of mefeat (f32 channel-major).
// ----------------------------------------------------------------------------
__global__ void corr_lookup_k(const float* __restrict__ flow,
                              const __bf16* c0, const __bf16* c1,
                              const __bf16* c2, const __bf16* c3,
                              float* __restrict__ mefeat) {
  const int HW = 8192;
  size_t idx = (size_t)blockIdx.x * blockDim.x + threadIdx.x;
  if (idx >= (size_t)HW * 324) return;
  int p = (int)(idx / 324), ch = (int)(idx % 324);
  int lev = ch / 81, j = ch % 81;
  float dy = (float)(j / 9) - 4.f, dx = (float)(j % 9) - 4.f;
  int x = p % 128, y = p / 128;
  float cx = (float)x + flow[p];
  float cy = (float)y + flow[HW + p];
  float s = (float)(1 << lev);
  float gx = cx / s + dy;          // reference: grid[...,0] = cent_x/s + dy
  float gy = cy / s + dx;          // reference: grid[...,1] = cent_y/s + dx
  int Hc = 64 >> lev, Wc = 128 >> lev;
  const __bf16* img = (lev == 0) ? c0 : (lev == 1) ? c1 : (lev == 2) ? c2 : c3;
  const __bf16* im = img + (size_t)p * Hc * Wc;
  float px = (gx + 1.f) * 0.5f * (float)(Wc - 1);
  float py = (gy + 1.f) * 0.5f * (float)(Hc - 1);
  px = fminf(fmaxf(px, 0.f), (float)(Wc - 1));
  py = fminf(fmaxf(py, 0.f), (float)(Hc - 1));
  int x0 = (int)floorf(px), y0 = (int)floorf(py);
  float wx = px - (float)x0, wy = py - (float)y0;
  int x1 = min(x0 + 1, Wc - 1), y1 = min(y0 + 1, Hc - 1);
  float v00 = bf2f(im[y0 * Wc + x0]);
  float v01 = bf2f(im[y0 * Wc + x1]);
  float v10 = bf2f(im[y1 * Wc + x0]);
  float v11 = bf2f(im[y1 * Wc + x1]);
  float v = v00 * (1 - wx) * (1 - wy) + v01 * wx * (1 - wy) +
            v10 * (1 - wx) * wy + v11 * wx * wy;
  mefeat[(size_t)(2 + ch) * HW + p] = v;
}

__global__ void flow_to_mefeat_k(const float* __restrict__ flow,
                                 float* __restrict__ mefeat) {
  int i = blockIdx.x * blockDim.x + threadIdx.x;
  if (i < 2 * 8192) mefeat[i] = flow[i];
}

// bilinear 2x upsample (half-pixel centers), 32ch 32x64 -> 64x128, f32
__global__ void resize2x_k(const float* __restrict__ in, float* __restrict__ out) {
  size_t idx = (size_t)blockIdx.x * blockDim.x + threadIdx.x;
  if (idx >= (size_t)32 * 8192) return;
  int c = (int)(idx / 8192), p = (int)(idx % 8192);
  int oy = p / 128, ox = p % 128;
  float sy = fminf(fmaxf((oy + 0.5f) * 0.5f - 0.5f, 0.f), 31.f);
  float sx = fminf(fmaxf((ox + 0.5f) * 0.5f - 0.5f, 0.f), 63.f);
  int y0 = (int)floorf(sy), x0 = (int)floorf(sx);
  int y1 = min(y0 + 1, 31), x1 = min(x0 + 1, 63);
  float wy = sy - y0, wx = sx - x0;
  const float* im = in + (size_t)c * 2048;
  float v = im[y0 * 64 + x0] * (1 - wx) * (1 - wy) +
            im[y0 * 64 + x1] * wx * (1 - wy) +
            im[y1 * 64 + x0] * (1 - wx) * wy +
            im[y1 * 64 + x1] * wx * wy;
  out[idx] = v;
}

// ----------------- small elementwise kernels -----------------
__global__ void ew_unary_k(float* dst, const float* src, size_t n, int op) {
  size_t i = (size_t)blockIdx.x * blockDim.x + threadIdx.x;
  if (i < n) dst[i] = act_apply(src[i], op);
}
__global__ void ew_mul_k(float* dst, const float* a, const float* b, size_t n) {
  size_t i = (size_t)blockIdx.x * blockDim.x + threadIdx.x;
  if (i < n) dst[i] = a[i] * b[i];
}
__global__ void ew_gru_k(float* net, const float* z, const float* q, size_t n) {
  size_t i = (size_t)blockIdx.x * blockDim.x + threadIdx.x;
  if (i < n) net[i] = (1.f - z[i]) * net[i] + z[i] * q[i];
}
__global__ void ew_add_relu_k(float* dst, const float* a, const float* b, size_t n) {
  size_t i = (size_t)blockIdx.x * blockDim.x + threadIdx.x;
  if (i < n) dst[i] = fmaxf(a[i] + b[i], 0.f);
}
__global__ void ew_zero_k(float* dst, size_t n) {
  size_t i = (size_t)blockIdx.x * blockDim.x + threadIdx.x;
  if (i < n) dst[i] = 0.f;
}

// head: delta = tanh(w4 . h + b4); flow0 += delta; store prediction
__global__ void head_k(const float* __restrict__ h, const float* __restrict__ w4,
                       const float* __restrict__ b4, float* __restrict__ flow,
                       float* __restrict__ out_it) {
  int p = blockIdx.x * blockDim.x + threadIdx.x;
  if (p >= 8192) return;
  float s = b4[0];
  for (int c = 0; c < 128; ++c) s += h[(size_t)c * 8192 + p] * w4[c];
  float f = flow[p] + tanhf(s);
  flow[p] = f;
  out_it[p] = f;
}

// ============================================================================
// Host orchestration
// ============================================================================
extern "C" void kernel_launch(void* const* d_in, const int* in_sizes, int n_in,
                              void* d_out, int out_size, void* d_ws, size_t ws_size,
                              hipStream_t stream) {
  (void)in_sizes; (void)n_in; (void)out_size; (void)ws_size;
  const int HW = 8192, H = 64, W = 128;
  auto F = [&](int i) { return (const float*)d_in[i]; };

  // ---- workspace bump allocator ----
  char* base = (char*)d_ws;
  size_t off = 0;
  auto alloc = [&](size_t bytes) -> void* {
    void* p = base + off;
    off += (bytes + 255) & ~(size_t)255;
    return p;
  };

  float*  bufF768 = (float*) alloc((size_t)768 * HW * 4);
  __bf16* fmap1   = (__bf16*)alloc((size_t)HW * 768 * 2);
  __bf16* fmap2   = (__bf16*)alloc((size_t)HW * 768 * 2);
  __bf16* corr0   = (__bf16*)alloc((size_t)HW * HW * 2);
  __bf16* corr1   = (__bf16*)alloc((size_t)HW * 2048 * 2);
  __bf16* corr2   = (__bf16*)alloc((size_t)HW * 512 * 2);
  __bf16* corr3   = (__bf16*)alloc((size_t)HW * 128 * 2);
  float*  ctxX    = (float*) alloc((size_t)128 * HW * 4);
  float*  ctxT1   = (float*) alloc((size_t)128 * HW * 4);
  float*  ctxT2   = (float*) alloc((size_t)128 * HW * 4);
  float*  cnet    = (float*) alloc((size_t)256 * HW * 4);
  float*  net     = (float*) alloc((size_t)128 * HW * 4);
  float*  inp     = (float*) alloc((size_t)128 * HW * 4);
  float*  flow    = (float*) alloc((size_t)2 * HW * 4);
  float*  mefeat  = (float*) alloc((size_t)326 * HW * 4);
  float*  me_a    = (float*) alloc((size_t)128 * 2048 * 4);
  float*  me_b    = (float*) alloc((size_t)64 * 2048 * 4);
  float*  me_c    = (float*) alloc((size_t)32 * 2048 * 4);
  float*  m_up    = (float*) alloc((size_t)32 * HW * 4);
  float*  zb      = (float*) alloc((size_t)128 * HW * 4);
  float*  rb      = (float*) alloc((size_t)128 * HW * 4);
  float*  qb      = (float*) alloc((size_t)128 * HW * 4);
  float*  rnet    = (float*) alloc((size_t)128 * HW * 4);
  float*  hb1     = (float*) alloc((size_t)256 * HW * 4);
  float*  hb2     = (float*) alloc((size_t)256 * HW * 4);
  float*  hb3     = (float*) alloc((size_t)128 * HW * 4);
  __bf16* Abuf    = (__bf16*)alloc((size_t)HW * 2624 * 2);

  auto packW = [&](const float* w, int N, int K, int Kp) -> const __bf16* {
    __bf16* dst = (__bf16*)alloc((size_t)N * Kp * 2);
    size_t n = (size_t)N * Kp;
    pack_w_k<<<dim3((unsigned)((n + 255) / 256)), 256, 0, stream>>>(w, dst, N, K, Kp);
    return dst;
  };
  auto gemm = [&](const __bf16* A, const __bf16* Bt, const float* bias,
                  float* Cf, __bf16* Cb, int M, int N, int K, int act) {
    dim3 g((unsigned)(M / 128), (unsigned)((N + 127) / 128));
    gemm_bf16_wmma<<<g, 256, 0, stream>>>(A, Bt, bias, Cf, Cb, M, N, K, act);
  };
  auto im2col = [&](const float* s0, int c0, const float* s1, int c1,
                    const float* s2, int c2, int h, int w, int stride, int Kp) {
    size_t n = (size_t)(h / stride) * (w / stride) * Kp;
    im2col3x3_k<<<dim3((unsigned)((n + 255) / 256)), 256, 0, stream>>>(
        s0, c0, s1, c1, s2, c2, h, w, stride, Kp, Abuf);
  };
  auto gn = [&](float* x, const float* g, const float* b, int C, int hw, int act) {
    groupnorm_k<<<8, 256, 0, stream>>>(x, x, g, b, C, hw, act);
  };
  auto ew1 = [&](float* dst, const float* src, size_t n, int op) {
    ew_unary_k<<<dim3((unsigned)((n + 255) / 256)), 256, 0, stream>>>(dst, src, n, op);
  };

  // ---- pack all weights to bf16 [N][Kp], Kp multiple of 64 (params are the
  //      jax pytree flatten order: ctx_blocks, ctx_in/out, gru_q/r/z, h*, me*, tf)
  const __bf16* Wc1[3], *Wc2[3];
  for (int b = 0; b < 3; ++b) {
    Wc1[b] = packW(F(4 + 6 * b + 0), 128, 1152, 1152);
    Wc2[b] = packW(F(4 + 6 * b + 1), 128, 1152, 1152);
  }
  const __bf16* Wci = packW(F(23), 128, 768, 768);
  const __bf16* Wco = packW(F(25), 256, 128, 128);
  const __bf16* Wgq = packW(F(27), 128, 2592, 2624);
  const __bf16* Wgr = packW(F(29), 128, 2592, 2624);
  const __bf16* Wgz = packW(F(31), 128, 2592, 2624);
  const __bf16* Wh1 = packW(F(33), 256, 1152, 1152);
  const __bf16* Wh2 = packW(F(37), 256, 2304, 2304);
  const __bf16* Wh3 = packW(F(41), 128, 2304, 2304);
  const __bf16* Wm1 = packW(F(45), 128, 2934, 2944);   // 326*9 padded to /64
  const __bf16* Wm2 = packW(F(49), 64, 1152, 1152);
  const __bf16* Wm3 = packW(F(53), 32, 576, 576);
  const __bf16* Wtf = packW(F(57), 768, 2304, 2304);

  // ---- feature fusion: tf (1x1, Cin=2304) + GN + relu + l2norm ----
  {
    size_t n = (size_t)HW * 2304;
    pack_rows_k<<<dim3((unsigned)((n + 255) / 256)), 256, 0, stream>>>(
        F(0), 768, F(1), 768, F(2), 768, HW, 2304, Abuf);
  }
  gemm(Abuf, Wtf, F(56), bufF768, nullptr, HW, 768, 2304, 0);
  gn(bufF768, F(59), F(58), 768, HW, 1);                       // tf_gn + relu
  l2norm_pack_k<<<32, 256, 0, stream>>>(bufF768, fmap1, 768, HW);
  l2norm_pack_k<<<32, 256, 0, stream>>>(F(3), fmap2, 768, HW);

  // ---- context net ----
  gemm(fmap1, Wci, F(22), ctxX, nullptr, HW, 128, 768, 1);     // ctx_in + relu
  for (int b = 0; b < 3; ++b) {
    im2col(ctxX, 128, nullptr, 0, nullptr, 0, H, W, 1, 1152);
    gemm(Abuf, Wc1[b], nullptr, ctxT1, nullptr, HW, 128, 1152, 0);
    gn(ctxT1, F(4 + 6 * b + 3), F(4 + 6 * b + 2), 128, HW, 1); // g1 + relu
    im2col(ctxT1, 128, nullptr, 0, nullptr, 0, H, W, 1, 1152);
    gemm(Abuf, Wc2[b], nullptr, ctxT2, nullptr, HW, 128, 1152, 0);
    gn(ctxT2, F(4 + 6 * b + 5), F(4 + 6 * b + 4), 128, HW, 0); // g2
    ew_add_relu_k<<<dim3((unsigned)(((size_t)128 * HW + 255) / 256)), 256, 0,
                    stream>>>(ctxX, ctxT2, ctxX, (size_t)128 * HW);
  }
  {
    size_t n = (size_t)HW * 128;
    pack_rows_k<<<dim3((unsigned)((n + 255) / 256)), 256, 0, stream>>>(
        ctxX, 128, nullptr, 0, nullptr, 0, HW, 128, Abuf);
  }
  gemm(Abuf, Wco, F(24), cnet, nullptr, HW, 256, 128, 0);      // ctx_out
  ew1(net, cnet, (size_t)128 * HW, 3);                         // tanh
  ew1(inp, cnet + (size_t)128 * HW, (size_t)128 * HW, 1);      // relu

  // ---- correlation volume (the big GEMM) + pyramid, stored bf16 ----
  gemm(fmap1, fmap2, nullptr, nullptr, corr0, HW, HW, 768, 0);
  avgpool2_k<<<dim3((unsigned)(((size_t)HW * 2048 + 255) / 256)), 256, 0, stream>>>(corr0, corr1, 64, 128);
  avgpool2_k<<<dim3((unsigned)(((size_t)HW * 512 + 255) / 256)), 256, 0, stream>>>(corr1, corr2, 32, 64);
  avgpool2_k<<<dim3((unsigned)(((size_t)HW * 128 + 255) / 256)), 256, 0, stream>>>(corr2, corr3, 16, 32);
  ew_zero_k<<<64, 256, 0, stream>>>(flow, (size_t)2 * HW);

  // ---- 8 recurrent update iterations ----
  for (int it = 0; it < 8; ++it) {
    flow_to_mefeat_k<<<64, 256, 0, stream>>>(flow, mefeat);
    corr_lookup_k<<<dim3((unsigned)(((size_t)HW * 324 + 255) / 256)), 256, 0,
                    stream>>>(flow, corr0, corr1, corr2, corr3, mefeat);

    // motion encoder (stride-2 then two 3x3 at 32x64)
    im2col(mefeat, 326, nullptr, 0, nullptr, 0, H, W, 2, 2944);
    gemm(Abuf, Wm1, F(44), me_a, nullptr, 2048, 128, 2944, 0);
    groupnorm_k<<<8, 256, 0, stream>>>(me_a, me_a, F(47), F(46), 128, 2048, 1);
    im2col(me_a, 128, nullptr, 0, nullptr, 0, 32, 64, 1, 1152);
    gemm(Abuf, Wm2, F(48), me_b, nullptr, 2048, 64, 1152, 0);
    groupnorm_k<<<8, 256, 0, stream>>>(me_b, me_b, F(51), F(50), 64, 2048, 1);
    im2col(me_b, 64, nullptr, 0, nullptr, 0, 32, 64, 1, 576);
    gemm(Abuf, Wm3, F(52), me_c, nullptr, 2048, 32, 576, 0);
    groupnorm_k<<<8, 256, 0, stream>>>(me_c, me_c, F(55), F(54), 32, 2048, 1);
    resize2x_k<<<dim3((unsigned)(((size_t)32 * HW + 255) / 256)), 256, 0, stream>>>(me_c, m_up);

    // GRU (concat fused into im2col, K padded 2592 -> 2624)
    im2col(net, 128, inp, 128, m_up, 32, H, W, 1, 2624);
    gemm(Abuf, Wgz, F(30), zb, nullptr, HW, 128, 2624, 2);     // sigmoid
    gemm(Abuf, Wgr, F(28), rb, nullptr, HW, 128, 2624, 2);     // sigmoid
    ew_mul_k<<<dim3((unsigned)(((size_t)128 * HW + 255) / 256)), 256, 0,
               stream>>>(rnet, rb, net, (size_t)128 * HW);
    im2col(rnet, 128, inp, 128, m_up, 32, H, W, 1, 2624);
    gemm(Abuf, Wgq, F(26), qb, nullptr, HW, 128, 2624, 3);     // tanh
    ew_gru_k<<<dim3((unsigned)(((size_t)128 * HW + 255) / 256)), 256, 0,
               stream>>>(net, zb, qb, (size_t)128 * HW);

    // head
    im2col(net, 128, nullptr, 0, nullptr, 0, H, W, 1, 1152);
    gemm(Abuf, Wh1, F(32), hb1, nullptr, HW, 256, 1152, 0);
    gn(hb1, F(35), F(34), 256, HW, 1);
    im2col(hb1, 256, nullptr, 0, nullptr, 0, H, W, 1, 2304);
    gemm(Abuf, Wh2, F(36), hb2, nullptr, HW, 256, 2304, 0);
    gn(hb2, F(39), F(38), 256, HW, 1);
    im2col(hb2, 256, nullptr, 0, nullptr, 0, H, W, 1, 2304);
    gemm(Abuf, Wh3, F(40), hb3, nullptr, HW, 128, 2304, 1);    // relu fused
    head_k<<<32, 256, 0, stream>>>(hb3, F(43), F(42), flow,
                                   (float*)d_out + (size_t)it * HW);
  }
}
